// Attention_35940286333273
// MI455X (gfx1250) — matrix operand order repaired and checked by reading, not compiled, not run on hardware
//
#include <hip/hip_runtime.h>
#include <hip/hip_bf16.h>
#include <math.h>

// Problem constants (B, T, C, H from reference)
#define Bsz 2
#define Tn  2048
#define Cn  1024
#define Hn  16
#define Dn  64

typedef __attribute__((ext_vector_type(16))) _Float16 v16h;
typedef __attribute__((ext_vector_type(8)))  _Float16 v8h;
typedef __attribute__((ext_vector_type(8)))  float    v8f;

// ---------------------------------------------------------------------------
// WMMA fragment helpers (layouts per CDNA5 ISA 7.12.2, wave32)
// ---------------------------------------------------------------------------

// A-matrix 16x32 f16, row-major source with leading dim `ld`.
// lane<16 : row=lane,    a[0..7]=K0..7,  a[8..15]=K16..23
// lane>=16: row=lane-16, a[0..7]=K8..15, a[8..15]=K24..31
__device__ __forceinline__ v16h frag_a_rowmajor(const _Float16* p, int ld, int lane) {
  int row = lane & 15;
  int kb  = (lane >> 4) << 3;          // 0 or 8
  const _Float16* base = p + (long)row * ld + kb;
  v8h lo = *(const v8h*)(base);
  v8h hi = *(const v8h*)(base + 16);
  v16h a;
#pragma unroll
  for (int i = 0; i < 8; ++i) { a[i] = lo[i]; a[i + 8] = hi[i]; }
  return a;
}

// B-matrix 32x16 f16 where B[k][n] = p[n*ld + k] (K contiguous in memory).
// lane<16 : col=lane,    b[0..7]=K0..7,  b[8..15]=K16..23
// lane>=16: col=lane-16, b[0..7]=K8..15, b[8..15]=K24..31
__device__ __forceinline__ v16h frag_b_kcontig(const _Float16* p, int ld, int lane) {
  int n  = lane & 15;
  int kb = (lane >> 4) << 3;           // 0 or 8
  const _Float16* base = p + (long)n * ld + kb;
  v8h lo = *(const v8h*)(base);
  v8h hi = *(const v8h*)(base + 16);
  v16h b;
#pragma unroll
  for (int i = 0; i < 8; ++i) { b[i] = lo[i]; b[i + 8] = hi[i]; }
  return b;
}

__device__ __forceinline__ v8f wmma_f16(v16h a, v16h b, v8f c) {
  return __builtin_amdgcn_wmma_f32_16x16x32_f16(false, a, false, b, (short)0, c,
                                                false, false);
}

// ---------------------------------------------------------------------------
// Kernel 1: xp = x + pe -> f16 ; weights -> f16
// ---------------------------------------------------------------------------
__global__ void prep_kernel(const float* __restrict__ x,
                            const float* __restrict__ wqkv,
                            const float* __restrict__ wproj,
                            const float* __restrict__ pe,
                            _Float16* __restrict__ xp,
                            _Float16* __restrict__ wq,
                            _Float16* __restrict__ wp) {
  const long N1 = (long)Bsz * Tn * Cn;       // xp
  const long N2 = 3L * Cn * Cn;              // w_qkv
  const long N3 = (long)Cn * Cn;             // w_proj
  long i = (long)blockIdx.x * blockDim.x + threadIdx.x;
  if (i >= N1 + N2 + N3) return;
  if (i < N1) {
    xp[i] = (_Float16)(x[i] + pe[i % ((long)Tn * Cn)]);
  } else if (i < N1 + N2) {
    long j = i - N1;
    wq[j] = (_Float16)wqkv[j];
  } else {
    long j = i - N1 - N2;
    wp[j] = (_Float16)wproj[j];
  }
}

// ---------------------------------------------------------------------------
// Kernel 2: qkv = xp @ w_qkv^T. One wave -> 32x64 output tile:
// 2 A fragments x 4 shared B fragments -> 8 WMMAs per k-step, with
// software-pipelined (double-buffered) fragment loads.
// Scatter into q/k (row-major [bh][t][64]) and v transposed ([bh][64][t]).
// ---------------------------------------------------------------------------
__global__ void qkv_gemm_kernel(const _Float16* __restrict__ xp,
                                const _Float16* __restrict__ w,
                                _Float16* __restrict__ qh,
                                _Float16* __restrict__ kh,
                                _Float16* __restrict__ vh) {
  const int NT = (3 * Cn) / 64;              // 48 n-tiles
  int wave = (int)((blockIdx.x * blockDim.x + threadIdx.x) >> 5);
  int lane = threadIdx.x & 31;
  int m0 = (wave / NT) * 32;
  int n0 = (wave % NT) * 64;

  const _Float16* A0 = xp + (long)m0 * Cn;
  const _Float16* A1 = A0 + (long)16 * Cn;

  // prologue loads (k=0)
  v16h a0 = frag_a_rowmajor(A0, Cn, lane);
  v16h a1 = frag_a_rowmajor(A1, Cn, lane);
  v16h bb[4];
#pragma unroll
  for (int g = 0; g < 4; ++g)
    bb[g] = frag_b_kcontig(w + (long)(n0 + g * 16) * Cn, Cn, lane);

  v8f acc[8] = {};
  for (int k0 = 0; k0 < Cn; k0 += 32) {
    int k1 = k0 + 32;
    if (k1 >= Cn) k1 = 0;                    // clamp: harmless redundant load
    // prefetch next k-step while current WMMAs execute
    v16h na0 = frag_a_rowmajor(A0 + k1, Cn, lane);
    v16h na1 = frag_a_rowmajor(A1 + k1, Cn, lane);
    v16h nb[4];
#pragma unroll
    for (int g = 0; g < 4; ++g)
      nb[g] = frag_b_kcontig(w + (long)(n0 + g * 16) * Cn + k1, Cn, lane);

#pragma unroll
    for (int g = 0; g < 4; ++g) {
      acc[g]     = wmma_f16(a0, bb[g], acc[g]);
      acc[4 + g] = wmma_f16(a1, bb[g], acc[4 + g]);
    }
    a0 = na0; a1 = na1;
#pragma unroll
    for (int g = 0; g < 4; ++g) bb[g] = nb[g];
  }

  int rbase = 8 * (lane >> 4);
  int col   = lane & 15;
#pragma unroll
  for (int rt = 0; rt < 2; ++rt) {
#pragma unroll
    for (int g = 0; g < 4; ++g) {
#pragma unroll
      for (int r = 0; r < 8; ++r) {
        int m = m0 + rt * 16 + rbase + r;
        int n = n0 + g * 16 + col;
        int b = m >> 11, t = m & (Tn - 1);     // Tn = 2048
        int sel = n >> 10, cc = n & (Cn - 1);  // Cn = 1024
        int h = cc >> 6, dd = cc & 63;
        int bh = b * Hn + h;
        _Float16 val = (_Float16)acc[rt * 4 + g][r];
        if (sel == 0)      qh[((long)bh * Tn + t) * Dn + dd] = val;
        else if (sel == 1) kh[((long)bh * Tn + t) * Dn + dd] = val;
        else               vh[((long)bh * Dn + dd) * Tn + t] = val;  // transposed
      }
    }
  }
}

// ---------------------------------------------------------------------------
// Kernel 3: causal flash attention. One wave = 16 query rows.
// S = Q K^T via 4 WMMAs / 32-key block; V fragments and next-block K
// fragments are issued before the softmax so VALU work hides their latency;
// P is restaged via LDS into the A-fragment layout; O += P V via 4 WMMAs.
// ---------------------------------------------------------------------------
__global__ void attn_kernel(const _Float16* __restrict__ qh,
                            const _Float16* __restrict__ kh,
                            const _Float16* __restrict__ vh,
                            _Float16* __restrict__ yh) {
  __shared__ __align__(16) _Float16 lds_p[4][16 * 32];
  int wid  = threadIdx.x >> 5;
  int lane = threadIdx.x & 31;
  int bh = blockIdx.x >> 5;                  // Tn/64 = 32 q-tiles per head
  int qt = blockIdx.x & 31;
  int q0 = qt * 64 + wid * 16;

  const _Float16* Q = qh + (long)bh * Tn * Dn;
  const _Float16* K = kh + (long)bh * Tn * Dn;
  const _Float16* V = vh + (long)bh * Dn * Tn;   // [64][Tn]

  v16h aq0 = frag_a_rowmajor(Q + (long)q0 * Dn,      Dn, lane);
  v16h aq1 = frag_a_rowmajor(Q + (long)q0 * Dn + 32, Dn, lane);

  v8f o[4] = {};
  float mrow[8], lrow[8];
  int rowq[8];
  int rbase = 8 * (lane >> 4);
  int colk  = lane & 15;
#pragma unroll
  for (int r = 0; r < 8; ++r) {
    mrow[r] = -1e30f; lrow[r] = 0.0f; rowq[r] = q0 + rbase + r;
  }
  _Float16* P = lds_p[wid];
  const int jlast = q0 + 15;

  // prologue: K fragments for block j0 = 0
  v16h bk[4];
  bk[0] = frag_b_kcontig(K, Dn, lane);
  bk[1] = frag_b_kcontig(K + 32, Dn, lane);
  bk[2] = frag_b_kcontig(K + (long)16 * Dn, Dn, lane);
  bk[3] = frag_b_kcontig(K + (long)16 * Dn + 32, Dn, lane);

  for (int j0 = 0; j0 <= jlast; j0 += 32) {
    v8f s0 = {}, s1 = {};
    s0 = wmma_f16(aq0, bk[0], s0);
    s0 = wmma_f16(aq1, bk[1], s0);
    s1 = wmma_f16(aq0, bk[2], s1);
    s1 = wmma_f16(aq1, bk[3], s1);

    // issue V (this block) and K (next block, clamped) loads now; the
    // softmax VALU work below hides their latency.
    v16h bv[4];
#pragma unroll
    for (int g = 0; g < 4; ++g)
      bv[g] = frag_b_kcontig(V + (long)(g * 16) * Tn + j0, Tn, lane);
    int jn = (j0 + 32 <= jlast) ? (j0 + 32) : j0;
    v16h nk[4];
    nk[0] = frag_b_kcontig(K + (long)jn * Dn, Dn, lane);
    nk[1] = frag_b_kcontig(K + (long)jn * Dn + 32, Dn, lane);
    nk[2] = frag_b_kcontig(K + (long)(jn + 16) * Dn, Dn, lane);
    nk[3] = frag_b_kcontig(K + (long)(jn + 16) * Dn + 32, Dn, lane);

    bool boundary = (j0 + 31 > q0);
    float f[8];
#pragma unroll
    for (int r = 0; r < 8; ++r) {
      float a0 = s0[r] * 0.125f;             // 1/sqrt(64)
      float a1 = s1[r] * 0.125f;
      if (boundary) {
        if (j0 + colk      > rowq[r]) a0 = -1e30f;
        if (j0 + 16 + colk > rowq[r]) a1 = -1e30f;
      }
      // row max over 16 lanes of this half-wave (C-layout: row fixed per half)
      float vm = fmaxf(a0, a1);
      vm = fmaxf(vm, __shfl_xor(vm, 1, 32));
      vm = fmaxf(vm, __shfl_xor(vm, 2, 32));
      vm = fmaxf(vm, __shfl_xor(vm, 4, 32));
      vm = fmaxf(vm, __shfl_xor(vm, 8, 32));
      float mn = fmaxf(mrow[r], vm);
      f[r] = __expf(mrow[r] - mn);
      mrow[r] = mn;
      float p0 = __expf(a0 - mn);
      float p1 = __expf(a1 - mn);
      float rs = p0 + p1;
      rs += __shfl_xor(rs, 1, 32);
      rs += __shfl_xor(rs, 2, 32);
      rs += __shfl_xor(rs, 4, 32);
      rs += __shfl_xor(rs, 8, 32);
      lrow[r] = lrow[r] * f[r] + rs;
      int row = rbase + r;                   // stage P (16x32) to LDS
      P[row * 32 + colk]      = (_Float16)p0;
      P[row * 32 + 16 + colk] = (_Float16)p1;
    }
#pragma unroll
    for (int g = 0; g < 4; ++g)
#pragma unroll
      for (int r = 0; r < 8; ++r) o[g][r] *= f[r];

    // ensure LDS stores complete before cross-lane fragment read
    asm volatile("s_wait_dscnt 0x0" ::: "memory");

    v16h ap;
    {
      int prow = lane & 15;
      int pkb  = (lane >> 4) << 3;
      v8h lo = *(const v8h*)(P + prow * 32 + pkb);
      v8h hi = *(const v8h*)(P + prow * 32 + pkb + 16);
#pragma unroll
      for (int i = 0; i < 8; ++i) { ap[i] = lo[i]; ap[i + 8] = hi[i]; }
    }
#pragma unroll
    for (int g = 0; g < 4; ++g)
      o[g] = wmma_f16(ap, bv[g], o[g]);

#pragma unroll
    for (int g = 0; g < 4; ++g) bk[g] = nk[g];
  }

  // epilogue: divide by row sums, write y as [B*T, C] f16 for proj GEMM
  int b = bh / Hn, h = bh % Hn;
#pragma unroll
  for (int r = 0; r < 8; ++r) {
    float inv = 1.0f / lrow[r];
    long m = (long)b * Tn + q0 + rbase + r;
#pragma unroll
    for (int g = 0; g < 4; ++g) {
      int cidx = h * Dn + g * 16 + colk;
      yh[m * Cn + cidx] = (_Float16)(o[g][r] * inv);
    }
  }
}

// ---------------------------------------------------------------------------
// Kernel 4: out = y @ w_proj^T (fp32 output), 32x64 tile per wave,
// double-buffered like the QKV GEMM.
// ---------------------------------------------------------------------------
__global__ void proj_gemm_kernel(const _Float16* __restrict__ yh,
                                 const _Float16* __restrict__ w,
                                 float* __restrict__ out) {
  const int NT = Cn / 64;                    // 16 n-tiles
  int wave = (int)((blockIdx.x * blockDim.x + threadIdx.x) >> 5);
  int lane = threadIdx.x & 31;
  int m0 = (wave / NT) * 32;
  int n0 = (wave % NT) * 64;

  const _Float16* A0 = yh + (long)m0 * Cn;
  const _Float16* A1 = A0 + (long)16 * Cn;

  v16h a0 = frag_a_rowmajor(A0, Cn, lane);
  v16h a1 = frag_a_rowmajor(A1, Cn, lane);
  v16h bb[4];
#pragma unroll
  for (int g = 0; g < 4; ++g)
    bb[g] = frag_b_kcontig(w + (long)(n0 + g * 16) * Cn, Cn, lane);

  v8f acc[8] = {};
  for (int k0 = 0; k0 < Cn; k0 += 32) {
    int k1 = k0 + 32;
    if (k1 >= Cn) k1 = 0;
    v16h na0 = frag_a_rowmajor(A0 + k1, Cn, lane);
    v16h na1 = frag_a_rowmajor(A1 + k1, Cn, lane);
    v16h nb[4];
#pragma unroll
    for (int g = 0; g < 4; ++g)
      nb[g] = frag_b_kcontig(w + (long)(n0 + g * 16) * Cn + k1, Cn, lane);

#pragma unroll
    for (int g = 0; g < 4; ++g) {
      acc[g]     = wmma_f16(a0, bb[g], acc[g]);
      acc[4 + g] = wmma_f16(a1, bb[g], acc[4 + g]);
    }
    a0 = na0; a1 = na1;
#pragma unroll
    for (int g = 0; g < 4; ++g) bb[g] = nb[g];
  }

  int rbase = 8 * (lane >> 4);
  int col   = lane & 15;
#pragma unroll
  for (int rt = 0; rt < 2; ++rt)
#pragma unroll
    for (int g = 0; g < 4; ++g)
#pragma unroll
      for (int r = 0; r < 8; ++r)
        out[(long)(m0 + rt * 16 + rbase + r) * Cn + n0 + g * 16 + col] =
            acc[rt * 4 + g][r];
}

// ---------------------------------------------------------------------------
extern "C" void kernel_launch(void* const* d_in, const int* in_sizes, int n_in,
                              void* d_out, int out_size, void* d_ws, size_t ws_size,
                              hipStream_t stream) {
  const float* x     = (const float*)d_in[0];
  const float* wqkv  = (const float*)d_in[1];
  const float* wproj = (const float*)d_in[2];
  const float* pe    = (const float*)d_in[3];
  float* out = (float*)d_out;

  // Workspace layout (f16): xp | wqkv16 | wproj16 | q | k | vT | y  = 48 MB
  _Float16* ws   = (_Float16*)d_ws;
  _Float16* xp16 = ws;
  _Float16* wq16 = xp16 + (size_t)Bsz * Tn * Cn;
  _Float16* wp16 = wq16 + (size_t)3 * Cn * Cn;
  _Float16* qh   = wp16 + (size_t)Cn * Cn;
  _Float16* kh   = qh + (size_t)Bsz * Hn * Tn * Dn;
  _Float16* vh   = kh + (size_t)Bsz * Hn * Tn * Dn;
  _Float16* yh   = vh + (size_t)Bsz * Hn * Tn * Dn;

  long total = (long)Bsz * Tn * Cn + 3L * Cn * Cn + (long)Cn * Cn;
  prep_kernel<<<(int)((total + 255) / 256), 256, 0, stream>>>(
      x, wqkv, wproj, pe, xp16, wq16, wp16);

  int qkv_waves = (Bsz * Tn / 32) * (3 * Cn / 64);   // 6144 waves
  qkv_gemm_kernel<<<qkv_waves * 32 / 256, 256, 0, stream>>>(xp16, wq16, qh, kh, vh);

  attn_kernel<<<Bsz * Hn * (Tn / 64), 128, 0, stream>>>(qh, kh, vh, yh);

  int proj_waves = (Bsz * Tn / 32) * (Cn / 64);      // 2048 waves
  proj_gemm_kernel<<<proj_waves * 32 / 256, 256, 0, stream>>>(yh, wp16, out);
}
